// EMPSN_motion_2774548873300
// MI455X (gfx1250) — matrix operand order repaired and checked by reading, not compiled, not run on hardware
//
#include <hip/hip_runtime.h>
#include <hip/hip_bf16.h>
#include <math.h>

#define HD 128
#define NPGc 200
#define N0c 20000
#define N1c 40000
#define N2c 20000
#define E00c 160000
#define E01c 80000
#define E11c 120000
#define E12c 60000
#define DINP_MSG 288
#define NW 2  // waves per block in WMMA kernels

typedef unsigned short u16;
typedef __attribute__((ext_vector_type(16))) __bf16 v16bf;
typedef __attribute__((ext_vector_type(16))) u16   v16u;
typedef __attribute__((ext_vector_type(8)))  float v8f;

union AU { unsigned int d[8]; v16u v; };
union BU { uint4 q[2]; v16u v; };

__device__ __forceinline__ float bf2f(u16 u) {
    unsigned int i = ((unsigned int)u) << 16;
    float f; __builtin_memcpy(&f, &i, 4); return f;
}
__device__ __forceinline__ u16 f2bf(float f) {
    unsigned int i; __builtin_memcpy(&i, &f, 4);
    unsigned int r = i + 0x7FFFu + ((i >> 16) & 1u);
    return (u16)(r >> 16);
}
__device__ __forceinline__ float siluf(float v) { return v / (1.f + __expf(-v)); }

__device__ __forceinline__ void acc_zero(v8f a[8]) {
#pragma unroll
    for (int nt = 0; nt < 8; ++nt)
#pragma unroll
        for (int j = 0; j < 8; ++j) a[nt][j] = 0.f;
}

// A from LDS row-tile (16 x dinp bf16), B from packed weights, software-pipelined.
// A 16x32 bf16 layout: lane m=L&15, half=L>>4; dword j holds K pair
//   k0 = kt*32 + half*16 + 2j.
// B packed: ((kt*8+nt)*32+lane)*16 + i  ==  W[kt*32+half*16+i][nt*16+(lane&15)]
__device__ __forceinline__ void gemm_lds(int lane, const u16* __restrict__ rowlds,
                                         int dinp, const u16* __restrict__ Wp,
                                         v8f acc[8]) {
    const int m = lane & 15, half = lane >> 4;
    const unsigned int* row32 = (const unsigned int*)rowlds;
    const int roww = dinp >> 1;
    const int nk = dinp >> 5;
    auto loadA = [&](int kt, AU& au) {
#pragma unroll
        for (int j = 0; j < 8; ++j) {
            int k0 = kt * 32 + half * 16 + 2 * j;
            au.d[j] = row32[m * roww + (k0 >> 1)];
        }
    };
    auto loadB = [&](int kt, int nt, BU& bu) {
        const uint4* wq = (const uint4*)(Wp + (((size_t)(kt * 8 + nt) * 32 + lane) << 4));
        bu.q[0] = wq[0]; bu.q[1] = wq[1];
    };
    AU a_cur, a_nxt;
    BU b_cur, b_nxt;
    loadA(0, a_cur);
    loadB(0, 0, b_cur);
    for (int kt = 0; kt < nk; ++kt) {
        if (kt + 1 < nk) loadA(kt + 1, a_nxt);
        v16bf A = __builtin_bit_cast(v16bf, a_cur.v);
#pragma unroll
        for (int nt = 0; nt < 8; ++nt) {
            if (nt < 7) loadB(kt, nt + 1, b_nxt);
            else if (kt + 1 < nk) loadB(kt + 1, 0, b_nxt);
            v16bf B = __builtin_bit_cast(v16bf, b_cur.v);
            acc[nt] = __builtin_amdgcn_wmma_f32_16x16x32_bf16(
                false, A, false, B, (short)0, acc[nt], false, false);
            b_cur = b_nxt;
        }
        a_cur = a_nxt;
    }
}

// C/D layout: lane L holds rows j+8*(L>>4), col nt*16+(L&15)
__device__ __forceinline__ void acc_to_hid(int lane, const v8f a[8], const float* b, u16* hid) {
    const int m = lane & 15, half = lane >> 4;
#pragma unroll
    for (int nt = 0; nt < 8; ++nt)
#pragma unroll
        for (int j = 0; j < 8; ++j) {
            int r = j + 8 * half, c = nt * 16 + m;
            hid[r * HD + c] = f2bf(siluf(a[nt][j] + b[c]));
        }
}
__device__ __forceinline__ void acc_to_f32(int lane, const v8f a[8], const float* b,
                                           float* mb, bool dosilu) {
    const int m = lane & 15, half = lane >> 4;
#pragma unroll
    for (int nt = 0; nt < 8; ++nt)
#pragma unroll
        for (int j = 0; j < 8; ++j) {
            int r = j + 8 * half, c = nt * 16 + m;
            float v = a[nt][j] + b[c];
            if (dosilu) v = siluf(v);
            mb[r * HD + c] = v;
        }
}

// ---------------- small utility kernels ----------------
__global__ void k_zero_f32(float* p, int n) {
    int t = blockIdx.x * blockDim.x + threadIdx.x;
    if (t < n) p[t] = 0.f;
}
// pack f32 weight [din x 128] into fragment order (zero-padded)
__global__ void k_pack_w(u16* dst, const float* src, int din, int total) {
    int t = blockIdx.x * blockDim.x + threadIdx.x;
    if (t >= total) return;
    int i = t & 15;
    int lane = (t >> 4) & 31;
    int nt = (t >> 9) & 7;
    int kt = t >> 12;
    int k = kt * 32 + (lane >> 4) * 16 + i;
    int n = nt * 16 + (lane & 15);
    dst[t] = (k < din) ? f2bf(src[k * HD + n]) : (u16)0;
}
__global__ void k_centroid(const float* loc, const float* vel, const int* xk,
                           const int* batchk, int nv, int N,
                           float* normv, float* pos0, float* v0) {
    int i = blockIdx.x * blockDim.x + threadIdx.x;
    if (i >= N) return;
    float vx = 0, vy = 0, vz = 0, px = 0, py = 0, pz = 0;
    for (int j = 0; j < nv; ++j) {
        int g = xk[i * nv + j] + NPGc * batchk[i];
        vx += vel[g * 3 + 0]; vy += vel[g * 3 + 1]; vz += vel[g * 3 + 2];
        px += loc[g * 3 + 0]; py += loc[g * 3 + 1]; pz += loc[g * 3 + 2];
    }
    float inv = 1.f / (float)nv;
    vx *= inv; vy *= inv; vz *= inv;
    normv[i] = sqrtf(vx * vx + vy * vy + vz * vz);
    if (pos0) {
        pos0[i * 3 + 0] = px * inv; pos0[i * 3 + 1] = py * inv; pos0[i * 3 + 2] = pz * inv;
        v0[i * 3 + 0] = vx; v0[i * 3 + 1] = vy; v0[i * 3 + 2] = vz;
    }
}
__global__ void k_embed(const float* embw, const float* embb, const float* normv,
                        float* x, u16* xh, int N) {
    int t = blockIdx.x * blockDim.x + threadIdx.x;
    if (t >= N * HD) return;
    int i = t >> 7, h = t & 127;
    float v = embw[h] * normv[i] + embb[h];
    x[t] = v; xh[t] = f2bf(v);
}
__global__ void k_dist(const float* pos0, const int* ia, const int* ib, int E, float* out) {
    int e = blockIdx.x * blockDim.x + threadIdx.x;
    if (e >= E) return;
    int a = ia[e], b = ib[e];
    float dx = pos0[a * 3] - pos0[b * 3];
    float dy = pos0[a * 3 + 1] - pos0[b * 3 + 1];
    float dz = pos0[a * 3 + 2] - pos0[b * 3 + 2];
    out[e] = sqrtf(dx * dx + dy * dy + dz * dz + 1e-12f);
}
__global__ void k_vec1(const float* pos0, const int* x1, const int* b1, float* vec1, int N) {
    int i = blockIdx.x * blockDim.x + threadIdx.x;
    if (i >= N) return;
    int g0 = x1[i * 2 + 0] + NPGc * b1[i];
    int g1 = x1[i * 2 + 1] + NPGc * b1[i];
    for (int c = 0; c < 3; ++c) vec1[i * 3 + c] = pos0[g0 * 3 + c] - pos0[g1 * 3 + c];
}
__global__ void k_inv11(const float* pos0, const float* vec1, const int* rows6,
                        const int* adj11, int E, float* out) {
    int e = blockIdx.x * blockDim.x + threadIdx.x;
    if (e >= E) return;
    for (int p = 0; p < 3; ++p) {
        int a = rows6[(2 * p) * E + e], b = rows6[(2 * p + 1) * E + e];
        float dx = pos0[a * 3] - pos0[b * 3];
        float dy = pos0[a * 3 + 1] - pos0[b * 3 + 1];
        float dz = pos0[a * 3 + 2] - pos0[b * 3 + 2];
        out[e * 4 + p] = sqrtf(dx * dx + dy * dy + dz * dz + 1e-12f);
    }
    int s = adj11[e], r = adj11[E + e];
    float sx = vec1[s * 3], sy = vec1[s * 3 + 1], sz = vec1[s * 3 + 2];
    float rx = vec1[r * 3], ry = vec1[r * 3 + 1], rz = vec1[r * 3 + 2];
    float dot = sx * rx + sy * ry + sz * rz;
    float ns = sqrtf(sx * sx + sy * sy + sz * sz);
    float nr = sqrtf(rx * rx + ry * ry + rz * rz);
    float c = dot / (ns * nr + 1e-12f);
    c = fminf(fmaxf(c, -1.f + 1e-6f), 1.f - 1e-6f);
    out[e * 4 + 3] = c;
}
__device__ __forceinline__ float angf(float dot, float nu, float nv) {
    float c = dot / (nu * nv + 1e-12f);
    c = fminf(fmaxf(c, -1.f + 1e-6f), 1.f - 1e-6f);
    return acosf(c);
}
__global__ void k_inv12(const float* pos0, const int* rows3, int E, float* out) {
    int e = blockIdx.x * blockDim.x + threadIdx.x;
    if (e >= E) return;
    int i1 = rows3[e], i2 = rows3[E + e], ia = rows3[2 * E + e];
    float p1x = pos0[i1 * 3], p1y = pos0[i1 * 3 + 1], p1z = pos0[i1 * 3 + 2];
    float p2x = pos0[i2 * 3], p2y = pos0[i2 * 3 + 1], p2z = pos0[i2 * 3 + 2];
    float ax = pos0[ia * 3], ay = pos0[ia * 3 + 1], az = pos0[ia * 3 + 2];
    float v1x = p1x - ax, v1y = p1y - ay, v1z = p1z - az;
    float v2x = p2x - ax, v2y = p2y - ay, v2z = p2z - az;
    float bx = p1x - p2x, by = p1y - p2y, bz = p1z - p2z;
    float n1 = sqrtf(v1x * v1x + v1y * v1y + v1z * v1z + 1e-12f);
    float n2 = sqrtf(v2x * v2x + v2y * v2y + v2z * v2z + 1e-12f);
    float nb = sqrtf(bx * bx + by * by + bz * bz + 1e-12f);
    float a1 = angf(v1x * bx + v1y * by + v1z * bz, n1, nb) +
               angf(v2x * bx + v2y * by + v2z * bz, n2, nb);
    float a2 = angf(v1x * v2x + v1y * v2y + v1z * v2z, n1, n2);
    out[e * 4 + 0] = n1; out[e * 4 + 1] = n2; out[e * 4 + 2] = a1; out[e * 4 + 3] = a2;
}
__global__ void k_cnt(const int* dst, int E, float* cnt) {
    int e = blockIdx.x * blockDim.x + threadIdx.x;
    if (e >= E) return;
    atomicAdd(&cnt[dst[e]], 1.f);
}

// ---------------- message kernel (WMMA) ----------------
__global__ __launch_bounds__(64) void k_msg(
    int E, int ninv, const int* __restrict__ src, const int* __restrict__ dst,
    const u16* __restrict__ xs, const u16* __restrict__ xr, const float* __restrict__ invf,
    const u16* __restrict__ W1p, const float* __restrict__ b1,
    const u16* __restrict__ W2p, const float* __restrict__ b2,
    const float* __restrict__ infw, const float* __restrict__ infb,
    float* __restrict__ mes, u16* __restrict__ m00h) {
    __shared__ __attribute__((aligned(16))) u16 s_rows[NW][16 * DINP_MSG];
    __shared__ __attribute__((aligned(16))) u16 s_hid[NW][16 * HD];
    __shared__ float s_m[NW][16 * HD];
    __shared__ int s_s[NW][16], s_d[NW][16];
    int w = threadIdx.x >> 5, lane = threadIdx.x & 31;
    int base = (blockIdx.x * NW + w) * 16;
    if (base >= E) return;
    u16* rows = s_rows[w]; u16* hid = s_hid[w]; float* mb = s_m[w];
    // prefetch indices for the 16 rows
    if (lane < 16) {
        int row = base + lane; if (row >= E) row = E - 1;
        s_s[w][lane] = src[row];
        s_d[w][lane] = dst[row];
    }
    asm volatile("s_wait_dscnt 0" ::: "memory");
    // stage gathered rows: [src feat 128 | dst feat 128 | inv ninv | zeros]
    for (int r = 0; r < 16; ++r) {
        int row = base + r; if (row >= E) row = E - 1;
        int sIdx = s_s[w][r], dIdx = s_d[w][r];
        const u16* p = (lane < 16) ? (xs + sIdx * HD + lane * 8)
                                   : (xr + dIdx * HD + (lane - 16) * 8);
        *(uint4*)(rows + r * DINP_MSG + lane * 8) = *(const uint4*)p;
        float iv = (lane < ninv) ? invf[row * ninv + lane] : 0.f;
        rows[r * DINP_MSG + 256 + lane] = f2bf(iv);
    }
    asm volatile("s_wait_dscnt 0" ::: "memory");
    v8f acc[8]; acc_zero(acc);
    gemm_lds(lane, rows, DINP_MSG, W1p, acc);
    acc_to_hid(lane, acc, b1, hid);
    asm volatile("s_wait_dscnt 0" ::: "memory");
    v8f acc2[8]; acc_zero(acc2);
    gemm_lds(lane, hid, HD, W2p, acc2);
    acc_to_f32(lane, acc2, b2, mb, true);  // message final SiLU
    asm volatile("s_wait_dscnt 0" ::: "memory");
    float ib = infb[0];
    for (int e = 0; e < 16; ++e) {
        int row = base + e;
        float p = 0.f;
#pragma unroll
        for (int i = 0; i < 4; ++i) { int h = lane + 32 * i; p += mb[e * HD + h] * infw[h]; }
        for (int o = 16; o > 0; o >>= 1) p += __shfl_xor(p, o);
        float wg = 1.f / (1.f + __expf(-(p + ib)));
        if (row < E) {
            int d = s_d[w][e];
#pragma unroll
            for (int i = 0; i < 4; ++i) {
                int h = lane + 32 * i;
                atomicAdd(&mes[d * HD + h], mb[e * HD + h] * wg);
            }
            if (m00h)
#pragma unroll
                for (int i = 0; i < 4; ++i) {
                    int h = lane + 32 * i;
                    m00h[row * HD + h] = f2bf(mb[e * HD + h]);
                }
        }
    }
}

// ---------------- update kernel (WMMA, residual, writes f32+bf16) ----------------
template <int F>
__global__ __launch_bounds__(64) void k_upd(
    int N, const u16* __restrict__ xh,
    const float* __restrict__ mA, const float* __restrict__ mB, const float* __restrict__ mC,
    const u16* __restrict__ W1p, const float* __restrict__ b1,
    const u16* __restrict__ W2p, const float* __restrict__ b2,
    float* __restrict__ xout, u16* __restrict__ xhout) {
    __shared__ __attribute__((aligned(16))) u16 s_rows[NW][16 * F * HD];
    __shared__ __attribute__((aligned(16))) u16 s_hid[NW][16 * HD];
    __shared__ float s_m[NW][16 * HD];
    int w = threadIdx.x >> 5, lane = threadIdx.x & 31;
    int base = (blockIdx.x * NW + w) * 16;
    if (base >= N) return;
    u16* rows = s_rows[w]; u16* hid = s_hid[w]; float* mb = s_m[w];
    const int dinp = F * HD;
    const float* parts[3] = {mA, mB, mC};
    for (int r = 0; r < 16; ++r) {
        int row = base + r; if (row >= N) row = N - 1;
        *(uint2*)(rows + r * dinp + lane * 4) = *(const uint2*)(xh + row * HD + lane * 4);
#pragma unroll
        for (int p = 1; p < F; ++p) {
            float4 f = *(const float4*)(parts[p - 1] + row * HD + lane * 4);
            uint2 o;
            o.x = (unsigned)f2bf(f.x) | ((unsigned)f2bf(f.y) << 16);
            o.y = (unsigned)f2bf(f.z) | ((unsigned)f2bf(f.w) << 16);
            *(uint2*)(rows + r * dinp + p * HD + lane * 4) = o;
        }
    }
    asm volatile("s_wait_dscnt 0" ::: "memory");
    v8f acc[8]; acc_zero(acc);
    gemm_lds(lane, rows, dinp, W1p, acc);
    acc_to_hid(lane, acc, b1, hid);
    asm volatile("s_wait_dscnt 0" ::: "memory");
    v8f acc2[8]; acc_zero(acc2);
    gemm_lds(lane, hid, HD, W2p, acc2);
    acc_to_f32(lane, acc2, b2, mb, false);
    asm volatile("s_wait_dscnt 0" ::: "memory");
    for (int e = 0; e < 16; ++e) {
        int row = base + e; if (row >= N) break;
#pragma unroll
        for (int i = 0; i < 4; ++i) {
            int h = lane + 32 * i;
            float v = xout[row * HD + h] + mb[e * HD + h];
            xout[row * HD + h] = v;
            xhout[row * HD + h] = f2bf(v);
        }
    }
}

// ---------------- pre['0'] kernel (WMMA, bf16 out) ----------------
__global__ __launch_bounds__(64) void k_pre(
    int N, const u16* __restrict__ xh,
    const u16* __restrict__ W1p, const float* __restrict__ b1,
    const u16* __restrict__ W2p, const float* __restrict__ b2, u16* __restrict__ outh) {
    __shared__ __attribute__((aligned(16))) u16 s_rows[NW][16 * HD];
    __shared__ __attribute__((aligned(16))) u16 s_hid[NW][16 * HD];
    __shared__ float s_m[NW][16 * HD];
    int w = threadIdx.x >> 5, lane = threadIdx.x & 31;
    int base = (blockIdx.x * NW + w) * 16;
    if (base >= N) return;
    u16* rows = s_rows[w]; u16* hid = s_hid[w]; float* mb = s_m[w];
    for (int r = 0; r < 16; ++r) {
        int row = base + r; if (row >= N) row = N - 1;
        *(uint2*)(rows + r * HD + lane * 4) = *(const uint2*)(xh + row * HD + lane * 4);
    }
    asm volatile("s_wait_dscnt 0" ::: "memory");
    v8f acc[8]; acc_zero(acc);
    gemm_lds(lane, rows, HD, W1p, acc);
    acc_to_hid(lane, acc, b1, hid);
    asm volatile("s_wait_dscnt 0" ::: "memory");
    v8f acc2[8]; acc_zero(acc2);
    gemm_lds(lane, hid, HD, W2p, acc2);
    acc_to_f32(lane, acc2, b2, mb, false);
    asm volatile("s_wait_dscnt 0" ::: "memory");
    for (int e = 0; e < 16; ++e) {
        int row = base + e; if (row >= N) break;
#pragma unroll
        for (int i = 0; i < 4; ++i) {
            int h = lane + 32 * i;
            outh[row * HD + h] = f2bf(mb[e * HD + h]);
        }
    }
}

// ---------------- post MLP + loss ----------------
__global__ __launch_bounds__(64) void k_postloss(
    int N, const u16* __restrict__ xh,
    const u16* __restrict__ W1p, const float* __restrict__ b1,
    const float* __restrict__ w2, const float* __restrict__ b2,
    const float* __restrict__ pos0, const float* __restrict__ loc,
    const float* __restrict__ y, float* __restrict__ out) {
    __shared__ __attribute__((aligned(16))) u16 s_rows[NW][16 * HD];
    __shared__ float s_m[NW][16 * HD];
    int w = threadIdx.x >> 5, lane = threadIdx.x & 31;
    int base = (blockIdx.x * NW + w) * 16;
    if (base >= N) return;
    u16* rows = s_rows[w]; float* mb = s_m[w];
    for (int r = 0; r < 16; ++r) {
        int row = base + r; if (row >= N) row = N - 1;
        *(uint2*)(rows + r * HD + lane * 4) = *(const uint2*)(xh + row * HD + lane * 4);
    }
    asm volatile("s_wait_dscnt 0" ::: "memory");
    v8f acc[8]; acc_zero(acc);
    gemm_lds(lane, rows, HD, W1p, acc);
    acc_to_f32(lane, acc, b1, mb, true);
    asm volatile("s_wait_dscnt 0" ::: "memory");
    float bb = b2[0];
    for (int e = 0; e < 16; ++e) {
        int row = base + e; if (row >= N) break;
        float p = 0.f;
#pragma unroll
        for (int i = 0; i < 4; ++i) { int h = lane + 32 * i; p += mb[e * HD + h] * w2[h]; }
        for (int o = 16; o > 0; o >>= 1) p += __shfl_xor(p, o);
        if (lane == 0) {
            float sc = p + bb, s = 0.f;
            for (int c = 0; c < 3; ++c) {
                float lp = sc * pos0[row * 3 + c] + loc[row * 3 + c] - y[row * 3 + c];
                s += lp * lp;
            }
            atomicAdd(out, s * (1.f / (float)(N * 3)));
        }
    }
}

// ---------------- EGNN position aggregation ----------------
__global__ __launch_bounds__(128) void k_trans(
    int E, const int* src, const int* dst, const u16* m00h,
    const float* posw, const float* posb, const float* pos0, float* agg) {
    int w = threadIdx.x >> 5, lane = threadIdx.x & 31;
    int e = blockIdx.x * 4 + w;
    if (e >= E) return;
    float p = 0.f;
#pragma unroll
    for (int i = 0; i < 4; ++i) { int h = lane + 32 * i; p += bf2f(m00h[e * HD + h]) * posw[h]; }
    for (int o = 16; o > 0; o >>= 1) p += __shfl_xor(p, o);
    float t = p + posb[0];
    int s = src[e], d = dst[e];
    if (lane < 3)
        atomicAdd(&agg[d * 3 + lane], (pos0[s * 3 + lane] - pos0[d * 3 + lane]) * t);
}
__global__ __launch_bounds__(128) void k_posvel(
    int N, const float* xf, const float* velw, const float* velb,
    const float* agg, const float* cnt, float* pos0, float* v0) {
    int w = threadIdx.x >> 5, lane = threadIdx.x & 31;
    int n = blockIdx.x * 4 + w;
    if (n >= N) return;
    float p = 0.f;
#pragma unroll
    for (int i = 0; i < 4; ++i) { int h = lane + 32 * i; p += xf[n * HD + h] * velw[h]; }
    for (int o = 16; o > 0; o >>= 1) p += __shfl_xor(p, o);
    float sc = p + velb[0];
    if (lane < 3) {
        float a = agg[n * 3 + lane] / fmaxf(cnt[n], 1.f);
        float vn = sc * v0[n * 3 + lane] + a;
        pos0[n * 3 + lane] += vn;
        v0[n * 3 + lane] = vn;
    }
}

static inline int cdiv(int a, int b) { return (a + b - 1) / b; }

extern "C" void kernel_launch(void* const* d_in, const int* in_sizes, int n_in,
                              void* d_out, int out_size, void* d_ws, size_t ws_size,
                              hipStream_t stream) {
    (void)in_sizes; (void)n_in; (void)out_size; (void)ws_size;
    const float* loc = (const float*)d_in[0];
    const float* vel = (const float*)d_in[1];
    const float* y   = (const float*)d_in[2];
    const int* x0i = (const int*)d_in[3];
    const int* x1i = (const int*)d_in[4];
    const int* x2i = (const int*)d_in[5];
    const int* b0i = (const int*)d_in[6];
    const int* b1i = (const int*)d_in[7];
    const int* b2i = (const int*)d_in[8];
    const int* adj00 = (const int*)d_in[9];
    const int* adj01 = (const int*)d_in[10];
    const int* adj11 = (const int*)d_in[11];
    const int* adj12 = (const int*)d_in[12];
    const int* i01 = (const int*)d_in[13];
    const int* i11 = (const int*)d_in[14];
    const int* i12 = (const int*)d_in[15];
    auto P = [&](int i) -> const float* { return (const float*)d_in[i]; };
    auto LB = [](int l) { return 18 + l * 52; };
    const int I_EMB_B = 16, I_EMB_W = 17;
    const int I_POST = 122, I_PRE0 = 126;

    char* wsp = (char*)d_ws;
    size_t cur = 0;
    auto alloc = [&](size_t bytes) -> void* {
        void* p = wsp + cur; cur += (bytes + 255) & ~(size_t)255; return p;
    };
    float* pos0 = (float*)alloc((size_t)N0c * 3 * 4);
    float* v0   = (float*)alloc((size_t)N0c * 3 * 4);
    float* norm0 = (float*)alloc((size_t)N0c * 4);
    float* norm1 = (float*)alloc((size_t)N1c * 4);
    float* norm2 = (float*)alloc((size_t)N2c * 4);
    float* x0f = (float*)alloc((size_t)N0c * HD * 4);
    float* x1f = (float*)alloc((size_t)N1c * HD * 4);
    float* x2f = (float*)alloc((size_t)N2c * HD * 4);
    u16* xh0 = (u16*)alloc((size_t)N0c * HD * 2);
    u16* xh1 = (u16*)alloc((size_t)N1c * HD * 2);
    u16* xh2 = (u16*)alloc((size_t)N2c * HD * 2);
    u16* x0ph = (u16*)alloc((size_t)N0c * HD * 2);
    float* mes00 = (float*)alloc((size_t)N0c * HD * 4);
    float* mes10 = (float*)alloc((size_t)N0c * HD * 4);
    float* mes01 = (float*)alloc((size_t)N1c * HD * 4);
    float* mes11 = (float*)alloc((size_t)N1c * HD * 4);
    float* mes21 = (float*)alloc((size_t)N1c * HD * 4);
    float* mes12 = (float*)alloc((size_t)N2c * HD * 4);
    u16* m00h = (u16*)alloc((size_t)E00c * HD * 2);
    float* vec1 = (float*)alloc((size_t)N1c * 3 * 4);
    float* inv00 = (float*)alloc((size_t)E00c * 4);
    float* inv01 = (float*)alloc((size_t)E01c * 4);
    float* inv11v = (float*)alloc((size_t)E11c * 4 * 4);
    float* inv12v = (float*)alloc((size_t)E12c * 4 * 4);
    float* agg = (float*)alloc((size_t)N0c * 3 * 4);
    float* cnt = (float*)alloc((size_t)N0c * 4);

    const int DIN[6] = {257, 257, 257, 260, 260, 260};
    const int Fd[3] = {3, 4, 2};
    const int Nd[3] = {N0c, N1c, N2c};
    float* xf[3] = {x0f, x1f, x2f};
    u16* xhh[3] = {xh0, xh1, xh2};

    u16 *Wmsg1[2][6], *Wmsg2[2][6], *Wupd1[2][3], *Wupd2[2][3];
    for (int l = 0; l < 2; ++l) {
        for (int t = 0; t < 6; ++t) {
            Wmsg1[l][t] = (u16*)alloc((size_t)9 * 4096 * 2);
            Wmsg2[l][t] = (u16*)alloc((size_t)4 * 4096 * 2);
        }
        for (int d = 0; d < 3; ++d) {
            Wupd1[l][d] = (u16*)alloc((size_t)Fd[d] * 4 * 4096 * 2);
            Wupd2[l][d] = (u16*)alloc((size_t)4 * 4096 * 2);
        }
    }
    u16* Wpre1 = (u16*)alloc((size_t)4 * 4096 * 2);
    u16* Wpre2 = (u16*)alloc((size_t)4 * 4096 * 2);
    u16* Wpost1 = (u16*)alloc((size_t)4 * 4096 * 2);

    auto pack = [&](u16* dstp, int idx, int din) {
        int total = ((din + 31) >> 5) * 4096;
        k_pack_w<<<cdiv(total, 256), 256, 0, stream>>>(dstp, P(idx), din, total);
    };
    auto zero = [&](float* p, int n) {
        k_zero_f32<<<cdiv(n, 256), 256, 0, stream>>>(p, n);
    };

    for (int l = 0; l < 2; ++l) {
        for (int t = 0; t < 6; ++t) {
            int Mb = LB(l) + t * 6;
            pack(Wmsg1[l][t], Mb + 3, DIN[t]);
            pack(Wmsg2[l][t], Mb + 5, HD);
        }
        for (int d = 0; d < 3; ++d) {
            int Ub = LB(l) + 38 + d * 4;
            pack(Wupd1[l][d], Ub + 1, Fd[d] * HD);
            pack(Wupd2[l][d], Ub + 3, HD);
        }
    }
    pack(Wpre1, I_PRE0 + 1, HD);
    pack(Wpre2, I_PRE0 + 3, HD);
    pack(Wpost1, I_POST + 1, HD);

    k_centroid<<<cdiv(N0c, 256), 256, 0, stream>>>(loc, vel, x0i, b0i, 1, N0c, norm0, pos0, v0);
    k_centroid<<<cdiv(N1c, 256), 256, 0, stream>>>(loc, vel, x1i, b1i, 2, N1c, norm1, nullptr, nullptr);
    k_centroid<<<cdiv(N2c, 256), 256, 0, stream>>>(loc, vel, x2i, b2i, 3, N2c, norm2, nullptr, nullptr);
    k_embed<<<cdiv(N0c * HD, 256), 256, 0, stream>>>(P(I_EMB_W), P(I_EMB_B), norm0, x0f, xh0, N0c);
    k_embed<<<cdiv(N1c * HD, 256), 256, 0, stream>>>(P(I_EMB_W), P(I_EMB_B), norm1, x1f, xh1, N1c);
    k_embed<<<cdiv(N2c * HD, 256), 256, 0, stream>>>(P(I_EMB_W), P(I_EMB_B), norm2, x2f, xh2, N2c);
    zero(cnt, N0c);
    k_cnt<<<cdiv(E00c, 256), 256, 0, stream>>>(adj00 + E00c, E00c, cnt);

    struct MsgCfg {
        const int *src, *dst; const u16 *xs, *xr; const float* inv;
        float* mes; int E, ninv, mesN;
    };
    MsgCfg mc[6] = {
        {adj00, adj00 + E00c, xh0, xh0, inv00, mes00, E00c, 1, N0c},
        {adj01, adj01 + E01c, xh0, xh1, inv01, mes01, E01c, 1, N1c},
        {adj01 + E01c, adj01, xh1, xh0, inv01, mes10, E01c, 1, N0c},
        {adj11, adj11 + E11c, xh1, xh1, inv11v, mes11, E11c, 4, N1c},
        {adj12, adj12 + E12c, xh1, xh2, inv12v, mes12, E12c, 4, N2c},
        {adj12 + E12c, adj12, xh2, xh1, inv12v, mes21, E12c, 4, N1c},
    };
    const float* partsA[3] = {mes00, mes01, mes12};
    const float* partsB[3] = {mes10, mes11, nullptr};
    const float* partsC[3] = {nullptr, mes21, nullptr};

    for (int l = 0; l < 2; ++l) {
        int L = LB(l);
        k_vec1<<<cdiv(N1c, 256), 256, 0, stream>>>(pos0, x1i, b1i, vec1, N1c);
        k_dist<<<cdiv(E00c, 256), 256, 0, stream>>>(pos0, adj00, adj00 + E00c, E00c, inv00);
        k_dist<<<cdiv(E01c, 256), 256, 0, stream>>>(pos0, i01, i01 + E01c, E01c, inv01);
        k_inv11<<<cdiv(E11c, 256), 256, 0, stream>>>(pos0, vec1, i11, adj11, E11c, inv11v);
        k_inv12<<<cdiv(E12c, 256), 256, 0, stream>>>(pos0, i12, E12c, inv12v);
        for (int t = 0; t < 6; ++t) zero(mc[t].mes, mc[t].mesN * HD);
        for (int t = 0; t < 6; ++t) {
            int Mb = L + t * 6;
            k_msg<<<cdiv(mc[t].E, 16 * NW), 64, 0, stream>>>(
                mc[t].E, mc[t].ninv, mc[t].src, mc[t].dst,
                mc[t].xs, mc[t].xr, mc[t].inv,
                Wmsg1[l][t], P(Mb + 2), Wmsg2[l][t], P(Mb + 4),
                P(Mb + 1), P(Mb + 0), mc[t].mes, (t == 0) ? m00h : nullptr);
        }
        {
            int Ub = L + 38 + 0 * 4;
            k_upd<3><<<cdiv(N0c, 16 * NW), 64, 0, stream>>>(
                N0c, xh0, partsA[0], partsB[0], partsC[0],
                Wupd1[l][0], P(Ub + 0), Wupd2[l][0], P(Ub + 2), x0f, xh0);
        }
        {
            int Ub = L + 38 + 1 * 4;
            k_upd<4><<<cdiv(N1c, 16 * NW), 64, 0, stream>>>(
                N1c, xh1, partsA[1], partsB[1], partsC[1],
                Wupd1[l][1], P(Ub + 0), Wupd2[l][1], P(Ub + 2), x1f, xh1);
        }
        {
            int Ub = L + 38 + 2 * 4;
            k_upd<2><<<cdiv(N2c, 16 * NW), 64, 0, stream>>>(
                N2c, xh2, partsA[2], partsB[2], partsC[2],
                Wupd1[l][2], P(Ub + 0), Wupd2[l][2], P(Ub + 2), x2f, xh2);
        }
        zero(agg, N0c * 3);
        k_trans<<<cdiv(E00c, 4), 128, 0, stream>>>(
            E00c, adj00, adj00 + E00c, m00h, P(L + 37), P(L + 36), pos0, agg);
        k_posvel<<<cdiv(N0c, 4), 128, 0, stream>>>(
            N0c, x0f, P(L + 51), P(L + 50), agg, cnt, pos0, v0);
    }

    k_pre<<<cdiv(N0c, 16 * NW), 64, 0, stream>>>(
        N0c, xh0, Wpre1, P(I_PRE0 + 0), Wpre2, P(I_PRE0 + 2), x0ph);
    k_zero_f32<<<1, 32, 0, stream>>>((float*)d_out, 1);
    k_postloss<<<cdiv(N0c, 16 * NW), 64, 0, stream>>>(
        N0c, x0ph, Wpost1, P(I_POST + 0), P(I_POST + 3), P(I_POST + 2),
        pos0, loc, y, (float*)d_out);
}
